// LinearAttentionLayer_42709154792029
// MI455X (gfx1250) — matrix operand order repaired and checked by reading, not compiled, not run on hardware
//
#include <hip/hip_runtime.h>
#include <cstdint>
#include <cstddef>

// ---------------- problem constants ----------------
#define BB   4
#define NN   4096
#define DD   1024
#define HH   16
#define HDD  64
#define FF   64
#define BN   (BB * NN)      // 16384 flat rows
#define NCHUNK 8            // split-K chunks for the KV reduction

// ---------------- types ----------------
typedef __attribute__((ext_vector_type(16))) __bf16       v16bf;
typedef __attribute__((ext_vector_type(8)))  float        v8f;
typedef __attribute__((ext_vector_type(4)))  unsigned int u32x4;
typedef __attribute__((ext_vector_type(4)))  float        f32x4;

union Frag { u32x4 u[2]; v16bf v; };

// ---------------- bf16 helpers (RNE) ----------------
__device__ __forceinline__ float bf2f(__bf16 h) {
  unsigned short s = __builtin_bit_cast(unsigned short, h);
  unsigned int u = ((unsigned int)s) << 16;
  return __builtin_bit_cast(float, u);
}
__device__ __forceinline__ __bf16 f2bf(float f) {
  unsigned int u = __builtin_bit_cast(unsigned int, f);
  unsigned int r = u + 0x7FFFu + ((u >> 16) & 1u);
  unsigned short s = (unsigned short)(r >> 16);
  return __builtin_bit_cast(__bf16, s);
}

// ---------------- conversion kernels ----------------
__global__ void k_f32_to_bf16_4(const float* __restrict__ in, __bf16* __restrict__ out) {
  size_t i = ((size_t)blockIdx.x * 256 + threadIdx.x) * 4;
  f32x4 v = *(const f32x4*)(in + i);
  out[i + 0] = f2bf(v.x);
  out[i + 1] = f2bf(v.y);
  out[i + 2] = f2bf(v.z);
  out[i + 3] = f2bf(v.w);
}

// Wt[n*dim + k] = W[k*dim + n]  (store W transposed, bf16)
__global__ void k_transpose_bf16(const float* __restrict__ W, __bf16* __restrict__ Wt, int dim) {
  int idx = blockIdx.x * 256 + threadIdx.x;
  if (idx >= dim * dim) return;
  int n = idx / dim;
  int k = idx - n * dim;
  Wt[idx] = f2bf(W[(size_t)k * dim + n]);
}

// ---------------- WMMA inner tile: one wave computes 32 x 64 ----------------
// A: row-major (M x K). Bt: B stored transposed (N x K, K-contiguous).
// All fragments for one k-step are loaded into distinct registers BEFORE the
// WMMA burst so the scheduler can issue staggered partial s_wait_loadcnt and
// overlap global loads with matrix math. Each B fragment feeds 2 WMMAs.
__device__ __forceinline__ void gemm_wave_32x64(const __bf16* __restrict__ Arow, int lda,
                                                const __bf16* __restrict__ Bt, int ldb,
                                                int K, v8f acc[2][4]) {
  const int lane = threadIdx.x & 31;
  const int lh   = lane & 15;
  const int hiA  = (lane >> 4) ? 8 : 0;   // A: high half-lanes take K+8 / K+24
  const int hiB  = (lane >> 4) ? 8 : 0;   // B: high half-lanes take K+8 / K+24
  const __bf16* ap0 = Arow + (size_t)lh * lda + hiA;
  const __bf16* ap1 = ap0 + (size_t)16 * lda;
#pragma unroll 2
  for (int k0 = 0; k0 < K; k0 += 32) {
    __builtin_prefetch(ap0 + k0 + 64, 0, 1);  // global_prefetch_b8
    Frag a0, a1, b[4];
    a0.u[0] = *(const u32x4*)(ap0 + k0);
    a0.u[1] = *(const u32x4*)(ap0 + k0 + 16);
    a1.u[0] = *(const u32x4*)(ap1 + k0);
    a1.u[1] = *(const u32x4*)(ap1 + k0 + 16);
#pragma unroll
    for (int j = 0; j < 4; ++j) {
      const __bf16* bp = Bt + (size_t)(j * 16 + lh) * ldb + k0 + hiB;
      b[j].u[0] = *(const u32x4*)(bp);
      b[j].u[1] = *(const u32x4*)(bp + 16);
    }
#pragma unroll
    for (int j = 0; j < 4; ++j) {
      acc[0][j] = __builtin_amdgcn_wmma_f32_16x16x32_bf16(false, a0.v, false, b[j].v,
                                                          (short)0, acc[0][j], false, false);
      acc[1][j] = __builtin_amdgcn_wmma_f32_16x16x32_bf16(false, a1.v, false, b[j].v,
                                                          (short)0, acc[1][j], false, false);
    }
  }
}

// ---------------- GEMM kernels (block = 256 thr = 8 waves, tile 256 x 64) ----------------

// C_bf16[M x 1024] = A @ W^T(Bt) + bias
__global__ void k_gemm_bias(const __bf16* __restrict__ A, const __bf16* __restrict__ Bt,
                            const float* __restrict__ bias, __bf16* __restrict__ C, int K) {
  const int w  = threadIdx.x >> 5;
  const int m0 = blockIdx.x * 256 + w * 32;
  const int n0 = blockIdx.y * 64;
  v8f acc[2][4] = {};
  gemm_wave_32x64(A + (size_t)m0 * DD, DD, Bt + (size_t)n0 * DD, DD, K, acc);
  const int lane = threadIdx.x & 31, lh = lane & 15;
  const int rbase = (lane >> 4) ? 8 : 0;
#pragma unroll
  for (int i = 0; i < 2; ++i)
#pragma unroll
    for (int j = 0; j < 4; ++j) {
      const int col = n0 + j * 16 + lh;
      const float bv = bias[col];
#pragma unroll
      for (int r = 0; r < 8; ++r) {
        const int row = m0 + i * 16 + rbase + r;
        C[(size_t)row * DD + col] = f2bf(acc[i][j][r] + bv);
      }
    }
}

// Per-head feature map: Qf[:, h*64+j] = elu(Q[:, h*64:+64] @ wf + bf) + 1
__global__ void k_gemm_fmap(const __bf16* __restrict__ Qin, const __bf16* __restrict__ wft,
                            const float* __restrict__ bias, __bf16* __restrict__ Qf) {
  const int w  = threadIdx.x >> 5;
  const int m0 = blockIdx.x * 256 + w * 32;
  const int h  = blockIdx.z;
  v8f acc[2][4] = {};
  gemm_wave_32x64(Qin + (size_t)m0 * DD + h * HDD, DD, wft, HDD, HDD, acc);
  const int lane = threadIdx.x & 31, lh = lane & 15;
  const int rbase = (lane >> 4) ? 8 : 0;
#pragma unroll
  for (int i = 0; i < 2; ++i)
#pragma unroll
    for (int j = 0; j < 4; ++j) {
      const int col = j * 16 + lh;  // 0..63 within head
      const float bv = bias[col];
#pragma unroll
      for (int r = 0; r < 8; ++r) {
        const int row = m0 + i * 16 + rbase + r;
        float c = acc[i][j][r] + bv;
        c = (c > 0.0f) ? (c + 1.0f) : __expf(c);  // elu(x)+1
        Qf[(size_t)row * DD + h * HDD + col] = f2bf(c);
      }
    }
}

// QKV: Obf[:, h] = (Qf_h @ KV_h) / (QZ + 1e-6)
__global__ void k_gemm_qkv(const __bf16* __restrict__ Qf, const __bf16* __restrict__ KVt,
                           const float* __restrict__ QZ, __bf16* __restrict__ Obf) {
  const int w  = threadIdx.x >> 5;
  const int m0 = blockIdx.x * 256 + w * 32;
  const int h  = blockIdx.z;
  const int b  = (blockIdx.x * 256) / NN;
  const int bh = b * HH + h;
  v8f acc[2][4] = {};
  gemm_wave_32x64(Qf + (size_t)m0 * DD + h * HDD, DD,
                  KVt + (size_t)bh * FF * HDD, FF, FF, acc);
  const int lane = threadIdx.x & 31, lh = lane & 15;
  const int rbase = (lane >> 4) ? 8 : 0;
#pragma unroll
  for (int i = 0; i < 2; ++i)
#pragma unroll
    for (int j = 0; j < 4; ++j) {
      const int col = j * 16 + lh;
#pragma unroll
      for (int r = 0; r < 8; ++r) {
        const int row = m0 + i * 16 + rbase + r;
        const float qz = QZ[(size_t)row * HH + h];
        Obf[(size_t)row * DD + h * HDD + col] = f2bf(acc[i][j][r] / (qz + 1e-6f));
      }
    }
}

// Final: Y = x + Obf @ Wo^T + bo   (f32, pre-LayerNorm, written to d_out)
__global__ void k_gemm_out(const __bf16* __restrict__ A, const __bf16* __restrict__ Bt,
                           const float* __restrict__ bias, const float* __restrict__ xres,
                           float* __restrict__ Y) {
  const int w  = threadIdx.x >> 5;
  const int m0 = blockIdx.x * 256 + w * 32;
  const int n0 = blockIdx.y * 64;
  v8f acc[2][4] = {};
  gemm_wave_32x64(A + (size_t)m0 * DD, DD, Bt + (size_t)n0 * DD, DD, DD, acc);
  const int lane = threadIdx.x & 31, lh = lane & 15;
  const int rbase = (lane >> 4) ? 8 : 0;
#pragma unroll
  for (int i = 0; i < 2; ++i)
#pragma unroll
    for (int j = 0; j < 4; ++j) {
      const int col = n0 + j * 16 + lh;
      const float bv = bias[col];
#pragma unroll
      for (int r = 0; r < 8; ++r) {
        const int row = m0 + i * 16 + rbase + r;
        const size_t idx = (size_t)row * DD + col;
        Y[idx] = xres[idx] + acc[i][j][r] + bv;
      }
    }
}

// ---------------- KV = Kf^T @ V per (b,h), split-K over N via LDS transpose ----------------
// grid (B*H, NCHUNK), block 128 (4 waves). Partials in f32 (deterministic).
__global__ void k_kv_partial(const __bf16* __restrict__ Kf, const __bf16* __restrict__ V,
                             float* __restrict__ part) {
  __shared__ __bf16 sKf[64 * 32];  // [f][n]
  __shared__ __bf16 sV [64 * 32];  // [hd][n]
  const int bh = blockIdx.x, b = bh / HH, h = bh % HH;
  const int chunk = blockIdx.y;
  const int tid = threadIdx.x, w = tid >> 5, lane = tid & 31, lh = lane & 15;
  const int hiA = (lane >> 4) ? 8 : 0;
  const int hiB = (lane >> 4) ? 8 : 0;
  v8f acc[4] = {};
  const int n_base = chunk * (NN / NCHUNK);
  for (int n0 = 0; n0 < NN / NCHUNK; n0 += 32) {
    // stage 32(n) x 64(cols) tiles, transposed into LDS
    {
      const int nloc = tid >> 2;       // 0..31
      const int cg   = tid & 3;        // 4 groups of 16 cols
      const size_t src = (size_t)(b * NN + n_base + n0 + nloc) * DD + h * HDD + cg * 16;
      Frag kf_, v_;
      kf_.u[0] = *(const u32x4*)(Kf + src);
      kf_.u[1] = *(const u32x4*)(Kf + src + 8);
      v_.u[0]  = *(const u32x4*)(V + src);
      v_.u[1]  = *(const u32x4*)(V + src + 8);
#pragma unroll
      for (int i = 0; i < 16; ++i) {
        sKf[(cg * 16 + i) * 32 + nloc] = ((const __bf16*)&kf_)[i];
        sV [(cg * 16 + i) * 32 + nloc] = ((const __bf16*)&v_)[i];
      }
    }
    __syncthreads();
    // wave w: rows f = w*16 .. w*16+15 of KV, all 64 hd columns.
    // Load A + all 4 B fragments first, then the WMMA burst.
    Frag a, bb[4];
    const __bf16* apx = &sKf[(w * 16 + lh) * 32 + hiA];
    a.u[0] = *(const u32x4*)(apx);
    a.u[1] = *(const u32x4*)(apx + 16);
#pragma unroll
    for (int j = 0; j < 4; ++j) {
      const __bf16* bp = &sV[(j * 16 + lh) * 32 + hiB];
      bb[j].u[0] = *(const u32x4*)(bp);
      bb[j].u[1] = *(const u32x4*)(bp + 16);
    }
#pragma unroll
    for (int j = 0; j < 4; ++j)
      acc[j] = __builtin_amdgcn_wmma_f32_16x16x32_bf16(false, a.v, false, bb[j].v,
                                                       (short)0, acc[j], false, false);
    __syncthreads();
  }
  float* out = part + ((size_t)bh * NCHUNK + chunk) * (FF * HDD);
  const int rbase = (lane >> 4) ? 8 : 0;
#pragma unroll
  for (int j = 0; j < 4; ++j) {
    const int hd = j * 16 + lh;
#pragma unroll
    for (int r = 0; r < 8; ++r) {
      const int f = w * 16 + rbase + r;
      out[f * HDD + hd] = acc[j][r];
    }
  }
}

// reduce NCHUNK partials, write KV transposed (KVt[bh][hd][f]) in bf16
__global__ void k_kv_reduce(const float* __restrict__ part, __bf16* __restrict__ KVt) {
  const int bh = blockIdx.x;
  for (int i = threadIdx.x; i < FF * HDD; i += 256) {
    const int f = i >> 6, hd = i & 63;
    float s = 0.0f;
#pragma unroll
    for (int c = 0; c < NCHUNK; ++c)
      s += part[((size_t)bh * NCHUNK + c) * (FF * HDD) + i];
    KVt[(size_t)bh * (FF * HDD) + hd * FF + f] = f2bf(s);
  }
}

// Z[bh][f] = sum_n Kf[b,n,h,f]
__global__ void k_zreduce(const __bf16* __restrict__ Kf, float* __restrict__ Z) {
  __shared__ float red[256];
  const int bh = blockIdx.x, b = bh / HH, h = bh % HH;
  const int f = threadIdx.x & 63, p = threadIdx.x >> 6;  // 4 N-partitions
  float s = 0.0f;
  const int span = NN / 4;
  for (int n = p * span; n < (p + 1) * span; ++n)
    s += bf2f(Kf[(size_t)(b * NN + n) * DD + h * HDD + f]);
  red[threadIdx.x] = s;
  __syncthreads();
  if (p == 0)
    Z[(size_t)bh * FF + f] = red[f] + red[64 + f] + red[128 + f] + red[192 + f];
}

// QZ[row, h] = dot(Qf[row, h*64:+64], Z[bh])
__global__ void k_qz(const __bf16* __restrict__ Qf, const float* __restrict__ Z,
                     float* __restrict__ QZ) {
  const int idx = blockIdx.x * 256 + threadIdx.x;  // BN*H total, exact
  const int row = idx / HH, h = idx % HH;
  const __bf16* q = Qf + (size_t)row * DD + h * HDD;
  const float* z = Z + ((size_t)(row / NN) * HH + h) * FF;
  float s = 0.0f;
#pragma unroll 8
  for (int f = 0; f < FF; ++f) s += bf2f(q[f]) * z[f];
  QZ[(size_t)row * HH + h] = s;
}

// ---------------- LayerNorm in-place on d_out (one block per row) ----------------
__global__ void k_layernorm(float* __restrict__ Y, const float* __restrict__ gamma,
                            const float* __restrict__ beta) {
  __shared__ float s1[256], s2[256];
  const int tid = threadIdx.x;
  float* y = Y + (size_t)blockIdx.x * DD;
  float v[4];
  float a = 0.0f, b = 0.0f;
#pragma unroll
  for (int i = 0; i < 4; ++i) {
    v[i] = y[tid + i * 256];
    a += v[i];
    b += v[i] * v[i];
  }
  s1[tid] = a; s2[tid] = b;
  __syncthreads();
  for (int o2 = 128; o2 > 0; o2 >>= 1) {
    if (tid < o2) { s1[tid] += s1[tid + o2]; s2[tid] += s2[tid + o2]; }
    __syncthreads();
  }
  const float mu  = s1[0] * (1.0f / DD);
  const float var = s2[0] * (1.0f / DD) - mu * mu;
  const float rs  = rsqrtf(var + 1e-5f);
#pragma unroll
  for (int i = 0; i < 4; ++i) {
    const int c = tid + i * 256;
    y[c] = (v[i] - mu) * rs * gamma[c] + beta[c];
  }
}

// ---------------- launch ----------------
extern "C" void kernel_launch(void* const* d_in, const int* in_sizes, int n_in,
                              void* d_out, int out_size, void* d_ws, size_t ws_size,
                              hipStream_t stream) {
  (void)in_sizes; (void)n_in; (void)out_size; (void)ws_size;
  const float* x     = (const float*)d_in[0];
  const float* wq    = (const float*)d_in[1];
  const float* bq    = (const float*)d_in[2];
  const float* wk    = (const float*)d_in[3];
  const float* bk    = (const float*)d_in[4];
  const float* wv    = (const float*)d_in[5];
  const float* bv    = (const float*)d_in[6];
  const float* wf    = (const float*)d_in[7];
  const float* bf_   = (const float*)d_in[8];
  const float* wo    = (const float*)d_in[9];
  const float* bo    = (const float*)d_in[10];
  const float* gamma = (const float*)d_in[11];
  const float* beta  = (const float*)d_in[12];

  char* ws = (char*)d_ws;
  size_t off = 0;
  auto take = [&](size_t bytes) -> char* {
    off = (off + 255) & ~(size_t)255;
    char* p = ws + off;
    off += bytes;
    return p;
  };

  __bf16* Xbf = (__bf16*)take((size_t)BN * DD * 2);
  __bf16* WtQ = (__bf16*)take((size_t)DD * DD * 2);
  __bf16* WtK = (__bf16*)take((size_t)DD * DD * 2);
  __bf16* WtV = (__bf16*)take((size_t)DD * DD * 2);
  __bf16* WtO = (__bf16*)take((size_t)DD * DD * 2);
  __bf16* wft = (__bf16*)take((size_t)HDD * FF * 2);
  __bf16* Qbf = (__bf16*)take((size_t)BN * DD * 2);
  __bf16* Kbf = (__bf16*)take((size_t)BN * DD * 2);
  __bf16* Vbf = (__bf16*)take((size_t)BN * DD * 2);
  __bf16* Qf  = (__bf16*)take((size_t)BN * DD * 2);
  float*  KVpart = (float*)take((size_t)BB * HH * NCHUNK * FF * HDD * 4);
  __bf16* KVt = (__bf16*)take((size_t)BB * HH * FF * HDD * 2);
  float*  Z   = (float*)take((size_t)BB * HH * FF * 4);
  float*  QZ  = (float*)take((size_t)BN * HH * 4);
  __bf16* Kf  = Xbf;  // reuse: Xbf dead after projections
  __bf16* Obf = Qbf;  // reuse: Qbf dead after feature map

  // 1) conversions
  k_f32_to_bf16_4<<<BN * DD / 1024, 256, 0, stream>>>(x, Xbf);
  k_transpose_bf16<<<(DD * DD) / 256, 256, 0, stream>>>(wq, WtQ, DD);
  k_transpose_bf16<<<(DD * DD) / 256, 256, 0, stream>>>(wk, WtK, DD);
  k_transpose_bf16<<<(DD * DD) / 256, 256, 0, stream>>>(wv, WtV, DD);
  k_transpose_bf16<<<(DD * DD) / 256, 256, 0, stream>>>(wo, WtO, DD);
  k_transpose_bf16<<<(HDD * FF) / 256, 256, 0, stream>>>(wf, wft, HDD);

  // 2) Q/K/V projections (WMMA bf16)
  dim3 blk(256);
  dim3 g1(BN / 256, DD / 64, 1);
  k_gemm_bias<<<g1, blk, 0, stream>>>(Xbf, WtQ, bq, Qbf, DD);
  k_gemm_bias<<<g1, blk, 0, stream>>>(Xbf, WtK, bk, Kbf, DD);
  k_gemm_bias<<<g1, blk, 0, stream>>>(Xbf, WtV, bv, Vbf, DD);

  // 3) per-head feature maps (WMMA + elu+1 epilogue)
  dim3 g2(BN / 256, 1, HH);
  k_gemm_fmap<<<g2, blk, 0, stream>>>(Qbf, wft, bf_, Qf);
  k_gemm_fmap<<<g2, blk, 0, stream>>>(Kbf, wft, bf_, Kf);

  // 4) Z and KV accumulation
  k_zreduce<<<BB * HH, 256, 0, stream>>>(Kf, Z);
  dim3 g3(BB * HH, NCHUNK, 1);
  k_kv_partial<<<g3, 128, 0, stream>>>(Kf, Vbf, KVpart);
  k_kv_reduce<<<BB * HH, 256, 0, stream>>>(KVpart, KVt);

  // 5) QZ and QKV (WMMA + divide epilogue)
  k_qz<<<(BN * HH) / 256, 256, 0, stream>>>(Qf, Z, QZ);
  dim3 g4(BN / 256, 1, HH);
  k_gemm_qkv<<<g4, blk, 0, stream>>>(Qf, KVt, QZ, Obf);

  // 6) output projection + bias + residual -> d_out (f32), then LayerNorm in place
  dim3 g5(BN / 256, DD / 64, 1);
  k_gemm_out<<<g5, blk, 0, stream>>>(Obf, WtO, bo, x, (float*)d_out);
  k_layernorm<<<BN, 256, 0, stream>>>((float*)d_out, gamma, beta);
}
